// stgp_33569464386041
// MI455X (gfx1250) — matrix operand order repaired and checked by reading, not compiled, not run on hardware
//
#include <hip/hip_runtime.h>

// ---------------------------------------------------------------------------
// Spatio-temporal GP predict, Kronecker-factored:
//   quad[t_te,s_te] = sum_{t1,t2} T[t_te,t1] T[t_te,t2] * M[t1,t2,s_te]
//   M[t1,t2,s_te]   = diag(S * Sigma_block(t1,t2) * S^T)[s_te]
// Sigma (400MB) is streamed exactly once; the 200x200-block GEMMs run on
// v_wmma_f32_16x16x32_bf16 with hi/lo bf16 split for ~fp32 accuracy.
// ---------------------------------------------------------------------------

typedef __attribute__((ext_vector_type(16))) __bf16 v16bf;
typedef __attribute__((ext_vector_type(8)))  float  v8f;

union Frag {
  v16bf v;
  uint4 q[2];
};

__device__ __forceinline__ unsigned short f32_bf16_rne(float x) {
  unsigned u = __float_as_uint(x);
  u += 0x7FFFu + ((u >> 16) & 1u);
  return (unsigned short)(u >> 16);
}
__device__ __forceinline__ float bf16_f32(unsigned short h) {
  return __uint_as_float(((unsigned)h) << 16);
}

// ---- phase 0: spatial SE kernel S[100][200] -------------------------------
__global__ void k_spatial(const float* __restrict__ tr_sp, const float* __restrict__ te_sp,
                          const float* __restrict__ ll_ls, const float* __restrict__ el_ls,
                          float* __restrict__ Sg) {
  int i = blockIdx.x * blockDim.x + threadIdx.x;
  if (i >= 100 * 200) return;
  int ste = i / 200, str = i - ste * 200;
  float ill = 1.0f / ll_ls[0];
  float iel = 1.0f / el_ls[0];
  float dx = (te_sp[ste * 3 + 0] - tr_sp[str * 3 + 0]) * ill;
  float dy = (te_sp[ste * 3 + 1] - tr_sp[str * 3 + 1]) * ill;
  float dz = (te_sp[ste * 3 + 2] - tr_sp[str * 3 + 2]) * iel;
  Sg[i] = expf(-(dx * dx + dy * dy + dz * dz));
}

// ---- phase 0: temporal SE kernel T[50][50] --------------------------------
__global__ void k_temporal(const float* __restrict__ tr_t, const float* __restrict__ te_t,
                           const float* __restrict__ tls, float* __restrict__ Tg) {
  int i = blockIdx.x * blockDim.x + threadIdx.x;
  if (i >= 50 * 50) return;
  int a = i / 50, b = i - a * 50;
  float inv = 1.0f / tls[0];
  float d = (te_t[a] - tr_t[b]) * inv;
  Tg[i] = expf(-d * d);
}

// ---- phase 0: W[t1][s_te(pad112)] = sum_s1 alpha[t1*200+s1] * S[s_te,s1] --
__global__ void k_alphaW(const float* __restrict__ alpha, const float* __restrict__ Sg,
                         float* __restrict__ W) {
  int i = blockIdx.x * blockDim.x + threadIdx.x;
  if (i >= 50 * 112) return;
  int t1 = i / 112, s = i - t1 * 112;
  float acc = 0.0f;
  if (s < 100) {
    const float* a = alpha + t1 * 200;
    const float* sr = Sg + s * 200;
    for (int k = 0; k < 200; ++k) acc += a[k] * sr[k];
  }
  W[i] = acc;
}

// ---- phase 1: per (t1,t2) block, M_vec = diag(S * B * S^T) via WMMA -------
// grid.x = 2500 (= t1*50+t2), block = 224 threads (7 waves, 1 m-tile each).
// LDS layout (u16 elements):
//   Shi [112][224] @ 0      Slo @ 25088
//   Bhi [208][224] @ 50176  Blo @ 96768   (transposed: [n=s2][k=s1])
__launch_bounds__(224, 1)
__global__ void k_blocks(const float* __restrict__ Sigma, const float* __restrict__ Sg,
                         float* __restrict__ Mout) {
  extern __shared__ unsigned short sm[];
  unsigned short* Shi = sm;
  unsigned short* Slo = sm + 25088;
  unsigned short* Bhi = sm + 50176;
  unsigned short* Blo = sm + 96768;

  const int tid = threadIdx.x;
  const int t1 = blockIdx.x / 50;
  const int t2 = blockIdx.x - t1 * 50;

  // Build S planes: thread tid owns column k = tid for every padded row m.
  {
    const int k = tid;
    for (int m = 0; m < 112; ++m) {
      float v = (m < 100 && k < 200) ? Sg[m * 200 + k] : 0.0f;
      unsigned short h = f32_bf16_rne(v);
      unsigned short l = f32_bf16_rne(v - bf16_f32(h));
      Shi[m * 224 + k] = h;
      Slo[m * 224 + k] = l;
    }
  }

  // Stream Sigma block (coalesced along s2), split to bf16, store transposed.
  {
    const float* blk = Sigma + (size_t)(t1 * 200) * 10000 + (size_t)(t2 * 200);
    for (int idx = tid; idx < 200 * 200; idx += 224) {
      int s1 = idx / 200;
      int s2 = idx - s1 * 200;
      float v = blk[(size_t)s1 * 10000 + s2];
      unsigned short h = f32_bf16_rne(v);
      unsigned short l = f32_bf16_rne(v - bf16_f32(h));
      int o = s2 * 224 + s1;  // [n][k]
      Bhi[o] = h;
      Blo[o] = l;
    }
    // zero-pad k in [200,224) for all n
    for (int idx = tid; idx < 208 * 24; idx += 224) {
      int n = idx / 24;
      int o = n * 224 + 200 + (idx - n * 24);
      Bhi[o] = 0; Blo[o] = 0;
    }
    // zero-pad n in [200,208) for k < 200
    for (int idx = tid; idx < 8 * 200; idx += 224) {
      int n = 200 + idx / 200;
      int o = n * 224 + (idx % 200);
      Bhi[o] = 0; Blo[o] = 0;
    }
  }
  __syncthreads();

  const int lane = tid & 31;
  const int mt = tid >> 5;        // wave id == m-tile (0..6)
  const int half = lane >> 4;     // 0: lanes 0-15, 1: lanes 16-31
  const int l15 = lane & 15;

  // A fragments (bf16 16x32 layout): lane M = l15;
  // element i -> K = k0 + (i<8 ? i : i+8) + (half ? 8 : 0)
  Frag ahi[7], alo[7];
  {
    const int mrow = mt * 16 + l15;
    const int koffA = half ? 8 : 0;
#pragma unroll
    for (int kk = 0; kk < 7; ++kk) {
      int base = mrow * 224 + kk * 32 + koffA;
      ahi[kk].q[0] = *(const uint4*)(Shi + base);
      ahi[kk].q[1] = *(const uint4*)(Shi + base + 16);
      alo[kk].q[0] = *(const uint4*)(Slo + base);
      alo[kk].q[1] = *(const uint4*)(Slo + base + 16);
    }
  }

  float pacc[8];
#pragma unroll
  for (int r = 0; r < 8; ++r) pacc[r] = 0.0f;

  // B fragments (bf16 32x16 layout): lane N = l15;
  // element i -> K = k0 + (i<8 ? i : i+8) + (half ? 8 : 0)
  const int koffB = half ? 8 : 0;
  for (int nt = 0; nt < 13; ++nt) {
    v8f c = {};
    const int nrow = nt * 16 + l15;
#pragma unroll
    for (int kk = 0; kk < 7; ++kk) {
      Frag bh, bl;
      int base = nrow * 224 + kk * 32 + koffB;
      bh.q[0] = *(const uint4*)(Bhi + base);
      bh.q[1] = *(const uint4*)(Bhi + base + 16);
      bl.q[0] = *(const uint4*)(Blo + base);
      bl.q[1] = *(const uint4*)(Blo + base + 16);
      // hi*hi + hi*lo + lo*hi  (lo*lo ~ 2^-18, dropped)
      c = __builtin_amdgcn_wmma_f32_16x16x32_bf16(false, ahi[kk].v, false, bh.v,
                                                  (short)0, c, false, false);
      c = __builtin_amdgcn_wmma_f32_16x16x32_bf16(false, ahi[kk].v, false, bl.v,
                                                  (short)0, c, false, false);
      c = __builtin_amdgcn_wmma_f32_16x16x32_bf16(false, alo[kk].v, false, bh.v,
                                                  (short)0, c, false, false);
    }
    // U[M][N] * S[M][N], accumulate per-row partial sums.
    // C layout: c[r] -> M = mt*16 + r + 8*half, N = nt*16 + l15
#pragma unroll
    for (int r = 0; r < 8; ++r) {
      int o = (mt * 16 + r + (half ? 8 : 0)) * 224 + (nt * 16 + l15);
      float s = bf16_f32(Shi[o]) + bf16_f32(Slo[o]);
      pacc[r] += c[r] * s;
    }
  }

  // Reduce across the 16 lanes of each half-wave (XOR butterflies stay inside).
#pragma unroll
  for (int r = 0; r < 8; ++r) {
    float v = pacc[r];
    v += __shfl_xor(v, 1);
    v += __shfl_xor(v, 2);
    v += __shfl_xor(v, 4);
    v += __shfl_xor(v, 8);
    pacc[r] = v;
  }
  if (l15 == 0) {
    float* dst = Mout + (size_t)blockIdx.x * 112 + mt * 16 + (half ? 8 : 0);
#pragma unroll
    for (int r = 0; r < 8; ++r) dst[r] = pacc[r];
  }
}

// ---- phase 2: fold temporal kernel; emit yPred / yVar ---------------------
// grid = 50 (t_te), block = 128 (s_te)
__global__ void k_phase2(const float* __restrict__ Tg, const float* __restrict__ W,
                         const float* __restrict__ Mout, const float* __restrict__ lsv,
                         float* __restrict__ out) {
  __shared__ float trow[64];
  const int t_te = blockIdx.x;
  const int s = threadIdx.x;
  if (s < 50) trow[s] = Tg[t_te * 50 + s];
  __syncthreads();
  if (s >= 100) return;

  float quad = 0.0f;
  for (int t1 = 0; t1 < 50; ++t1) {
    float w1 = trow[t1];
    float inner = 0.0f;
    const float* Mp = Mout + (size_t)(t1 * 50) * 112 + s;
    for (int t2 = 0; t2 < 50; ++t2) inner += trow[t2] * Mp[(size_t)t2 * 112];
    quad += w1 * inner;
  }
  float yp = 0.0f;
  for (int t1 = 0; t1 < 50; ++t1) yp += trow[t1] * W[t1 * 112 + s];

  // yPred.reshape(T_te,S_te).T -> out[s*50 + t]
  out[s * 50 + t_te] = yp;
  out[5000 + s * 50 + t_te] = expf(lsv[0]) - quad;
}

// ---------------------------------------------------------------------------
extern "C" void kernel_launch(void* const* d_in, const int* in_sizes, int n_in,
                              void* d_out, int out_size, void* d_ws, size_t ws_size,
                              hipStream_t stream) {
  const float* tr_sp = (const float*)d_in[0];
  const float* tr_t  = (const float*)d_in[1];
  const float* te_sp = (const float*)d_in[2];
  const float* te_t  = (const float*)d_in[3];
  const float* Sigma = (const float*)d_in[4];
  const float* alpha = (const float*)d_in[5];
  const float* ll_ls = (const float*)d_in[6];
  const float* el_ls = (const float*)d_in[7];
  const float* tm_ls = (const float*)d_in[8];
  const float* lsv   = (const float*)d_in[9];
  float* out = (float*)d_out;

  float* ws = (float*)d_ws;
  float* Sg = ws;            // 100*200       = 20000
  float* Tg = ws + 20000;    // 50*50         = 2500
  float* Wg = ws + 22500;    // 50*112        = 5600
  float* Mg = ws + 28100;    // 2500*112      = 280000  (~1.2 MB total)

  hipLaunchKernelGGL(k_spatial,  dim3((20000 + 255) / 256), dim3(256), 0, stream,
                     tr_sp, te_sp, ll_ls, el_ls, Sg);
  hipLaunchKernelGGL(k_temporal, dim3((2500 + 255) / 256), dim3(256), 0, stream,
                     tr_t, te_t, tm_ls, Tg);
  hipLaunchKernelGGL(k_alphaW,   dim3((5600 + 255) / 256), dim3(256), 0, stream,
                     alpha, Sg, Wg);
  hipLaunchKernelGGL(k_blocks,   dim3(2500), dim3(224), 286720, stream,
                     Sigma, Sg, Mg);
  hipLaunchKernelGGL(k_phase2,   dim3(50), dim3(128), 0, stream,
                     Tg, Wg, Mg, lsv, out);
}
